// FlowAttention_24584392803024
// MI455X (gfx1250) — matrix operand order repaired and checked by reading, not compiled, not run on hardware
//
#include <hip/hip_runtime.h>
#include <hip/hip_bf16.h>
#include <math.h>

typedef __attribute__((ext_vector_type(16))) __bf16 v16bf;
typedef __attribute__((ext_vector_type(8)))  float  v8f;

#define DIM   256
#define CTX   512
#define QLEN  64
#define BSZ   8
#define NEG_SLOPE 0.01f

// ---------------- workspace layout (floats) ----------------
// hp   : [8*512*256]  = 1048576   offset 0
// up   : [8*64*256]   = 131072    offset 1048576   (includes +b1)
// smax : [8*512]      = 4096      offset 1179648
#define WS_HP    0
#define WS_UP    1048576
#define WS_SMAX  1179648

// =====================================================================
// Kernel 1: hp = H @ Wh ; up = U @ Wu + b1   (0.6 GFLOP, VALU is fine)
// grid: 8*512 + 8*64 = 4608 blocks of 256 threads (thread = output e)
// =====================================================================
__global__ void flow_precompute(const float* __restrict__ H,
                                const float* __restrict__ U,
                                const float* __restrict__ W1,
                                const float* __restrict__ b1,
                                float* __restrict__ ws) {
    __shared__ float rowbuf[DIM];
    const int row = blockIdx.x;
    const int e   = threadIdx.x;

    const float* in;
    const float* W;
    float* outp;
    float acc;
    if (row < BSZ * CTX) {                 // hp rows
        in   = H + (size_t)row * DIM;
        W    = W1;                          // Wh = W1[0:256]
        outp = ws + WS_HP + (size_t)row * DIM;
        acc  = 0.0f;
    } else {                                // up rows
        int r2 = row - BSZ * CTX;
        in   = U + (size_t)r2 * DIM;
        W    = W1 + DIM * DIM;              // Wu = W1[256:512]
        outp = ws + WS_UP + (size_t)r2 * DIM;
        acc  = b1[e];
    }
    rowbuf[e] = in[e];
    __syncthreads();
    #pragma unroll 8
    for (int d = 0; d < DIM; ++d)
        acc = fmaf(rowbuf[d], W[d * DIM + e], acc);
    outp[e] = acc;
}

// =====================================================================
// Kernel 2: fused S / softmax-over-q / U_hat, bf16 WMMA for the big GEMM
// grid: (32 c-tiles, 8 batch), block = 256 threads = 8 waves.
// Wave w handles q in [w*8, w*8+8), processed TWO at a time so every
// B fragment load from LDS feeds two independent WMMA chains.
// __launch_bounds__(256, 1): the block is only 2 waves/SIMD anyway, so
// let the compiler use a big VGPR budget (keeps all 128 A-fragment
// VGPRs resident -- no scratch spills in the WMMA loop).
// LDS (dynamic, ~228KB of the WGP's 320KB):
//   wmF  : Wm as pre-swizzled B fragments  (bf16[128*512])  131072 B
//   ldsU : U[b]  f32 [64][256]                               65536 B
//   ldsH : H tile f32 [16][256]                              16384 B
//   ldsHp: hp tile f32 [16][256]                             16384 B
//   ldsS : S tile f32 [16][64]                                4096 B
// =====================================================================
__global__ void __launch_bounds__(256, 1)
flow_main(const float* __restrict__ H,
          const float* __restrict__ U,
          const float* __restrict__ W1,
          const float* __restrict__ W2,
          const float* __restrict__ b2,
          const int*   __restrict__ cmask,
          const int*   __restrict__ qmask,
          float* __restrict__ ws,
          float* __restrict__ out) {
    extern __shared__ __align__(16) char smem[];
    __bf16* wmF  = (__bf16*)(smem);
    float*  ldsU = (float*)(smem + 131072);
    float*  ldsH = (float*)(smem + 131072 + 65536);
    float*  ldsHp= (float*)(smem + 131072 + 65536 + 16384);
    float*  ldsS = (float*)(smem + 131072 + 65536 + 16384 + 16384);

    const int tid    = threadIdx.x;
    const int c_tile = blockIdx.x;      // 0..31
    const int b      = blockIdx.y;      // 0..7
    const float* hp  = ws + WS_HP;
    const float* up  = ws + WS_UP;

    // ---- stage U[b] (f32) ----
    for (int idx = tid; idx < QLEN * DIM; idx += 256)
        ldsU[idx] = U[(size_t)b * QLEN * DIM + idx];
    // ---- stage H tile + hp tile ----
    for (int idx = tid; idx < 16 * DIM; idx += 256) {
        size_t g = ((size_t)b * CTX + c_tile * 16) * DIM + idx;
        ldsH[idx]  = H[g];
        ldsHp[idx] = hp[g];
    }
    // ---- stage Wm swizzled into B-fragment layout ----
    // fragment f = etile*8 + ktile; within: lane*16 + i
    // B 32x16 bf16 layout: col = lane%16, K = (i/8)*16 + (lane/16)*8 + (i%8)
    for (int idx = tid; idx < 128 * 512; idx += 256) {
        int frag  = idx >> 9;
        int etile = frag >> 3, ktile = frag & 7;
        int wi    = idx & 511;
        int lane  = wi >> 4, i = wi & 15;
        int K     = ktile * 32 + ((i >> 3) << 4) + ((lane >> 4) << 3) + (i & 7);
        int e     = etile * 16 + (lane & 15);
        wmF[idx]  = (__bf16)W1[(size_t)(2 * DIM + K) * DIM + e];
    }
    __syncthreads();

    const int wave = tid >> 5;
    const int lane = tid & 31;
    const int r    = lane & 15;   // A-matrix row within c-tile
    const int h    = lane >> 4;   // half selector
    const int n    = lane & 15;   // D-matrix column (e within e-tile)
    const float b2v = b2[0];

    for (int jp = 0; jp < 4; ++jp) {
        const int q0 = wave * 8 + jp * 2;
        const int q1 = q0 + 1;

        // Build bf16 A fragments for BOTH q values (A = H row (*) U[q]).
        // 16-bit A 16x32 layout: K = (i/8)*16 + h*8 + (i%8)
        v16bf afr0[8], afr1[8];
        #pragma unroll
        for (int kt = 0; kt < 8; ++kt) {
            #pragma unroll
            for (int i = 0; i < 16; ++i) {
                int K = kt * 32 + ((i >> 3) << 4) + (h << 3) + (i & 7);
                float hv = ldsH[r * DIM + K];
                afr0[kt][i] = (__bf16)(hv * ldsU[q0 * DIM + K]);
                afr1[kt][i] = (__bf16)(hv * ldsU[q1 * DIM + K]);
            }
        }

        float sacc0[8], sacc1[8];
        #pragma unroll
        for (int v = 0; v < 8; ++v) { sacc0[v] = 0.0f; sacc1[v] = 0.0f; }

        for (int et = 0; et < 16; ++et) {
            v8f acc0 = {0.f, 0.f, 0.f, 0.f, 0.f, 0.f, 0.f, 0.f};
            v8f acc1 = {0.f, 0.f, 0.f, 0.f, 0.f, 0.f, 0.f, 0.f};
            #pragma unroll
            for (int kt = 0; kt < 8; ++kt) {
                // One B fragment load feeds two independent WMMA chains.
                const v16bf bfr =
                    *reinterpret_cast<const v16bf*>(wmF + ((et * 8 + kt) << 9) + (lane << 4));
                acc0 = __builtin_amdgcn_wmma_f32_16x16x32_bf16(
                           false, afr0[kt], false, bfr, (short)0, acc0, false, false);
                acc1 = __builtin_amdgcn_wmma_f32_16x16x32_bf16(
                           false, afr1[kt], false, bfr, (short)0, acc1, false, false);
            }
            const int   e    = et * 16 + n;
            const float w2v  = W2[e];
            const float upv0 = up[((size_t)b * QLEN + q0) * DIM + e];
            const float upv1 = up[((size_t)b * QLEN + q1) * DIM + e];
            #pragma unroll
            for (int v = 0; v < 8; ++v) {
                int   m   = v + 8 * h;                       // D-layout row
                float hpe = ldsHp[m * DIM + e];
                float v0  = acc0[v] + hpe + upv0;
                float v1  = acc1[v] + hpe + upv1;
                v0 = v0 > 0.0f ? v0 : NEG_SLOPE * v0;        // leaky_relu #1
                v1 = v1 > 0.0f ? v1 : NEG_SLOPE * v1;
                sacc0[v] = fmaf(v0, w2v, sacc0[v]);
                sacc1[v] = fmaf(v1, w2v, sacc1[v]);
            }
        }
        // reduce over the 16 N-lanes (stay inside each 16-lane half)
        #pragma unroll
        for (int off = 1; off < 16; off <<= 1) {
            #pragma unroll
            for (int v = 0; v < 8; ++v) {
                sacc0[v] += __shfl_xor(sacc0[v], off, 32);
                sacc1[v] += __shfl_xor(sacc1[v], off, 32);
            }
        }
        if (n == 0) {
            const bool qb0 = (qmask[b * QLEN + q0] == 0);
            const bool qb1 = (qmask[b * QLEN + q1] == 0);
            #pragma unroll
            for (int v = 0; v < 8; ++v) {
                int  m = v + 8 * h;
                int  c = c_tile * 16 + m;
                bool cb = (cmask[b * CTX + c] == 0);
                float s0 = sacc0[v] + b2v;
                float s1 = sacc1[v] + b2v;
                s0 = s0 > 0.0f ? s0 : NEG_SLOPE * s0;        // leaky_relu #2
                s1 = s1 > 0.0f ? s1 : NEG_SLOPE * s1;
                if (qb0 || cb) s0 = -1.0e9f;
                if (qb1 || cb) s1 = -1.0e9f;
                ldsS[m * QLEN + q0] = s0;
                ldsS[m * QLEN + q1] = s1;
            }
        }
    }
    __syncthreads();

    // softmax over q per context row; record Smax
    if (tid < 16) {
        float mx = -INFINITY;
        for (int q = 0; q < QLEN; ++q) mx = fmaxf(mx, ldsS[tid * QLEN + q]);
        ws[WS_SMAX + (size_t)b * CTX + c_tile * 16 + tid] = mx;
        float sum = 0.0f;
        for (int q = 0; q < QLEN; ++q) {
            float ev = __expf(ldsS[tid * QLEN + q] - mx);
            ldsS[tid * QLEN + q] = ev;
            sum += ev;
        }
        float inv = 1.0f / sum;
        for (int q = 0; q < QLEN; ++q) ldsS[tid * QLEN + q] *= inv;
    }
    __syncthreads();

    // U_hat[c,:] = A[c,:] @ U[b]   (16 rows x 256 cols, 256 threads)
    const int crow  = tid >> 4;
    const int ebase = (tid & 15) * 16;
    float accv[16];
    #pragma unroll
    for (int i = 0; i < 16; ++i) accv[i] = 0.0f;
    for (int q = 0; q < QLEN; ++q) {
        const float a = ldsS[crow * QLEN + q];
        #pragma unroll
        for (int i = 0; i < 16; ++i)
            accv[i] = fmaf(a, ldsU[q * DIM + ebase + i], accv[i]);
    }
    {
        const size_t UHAT_OFF = (size_t)BSZ * CTX * DIM;
        size_t base = UHAT_OFF + ((size_t)b * CTX + c_tile * 16 + crow) * DIM + ebase;
        #pragma unroll
        for (int i = 0; i < 16; ++i) out[base + i] = accv[i];
    }
}

// =====================================================================
// Kernel 3: b_attn = softmax_c(Smax); q2c = b_attn @ H; broadcast H_hat
// grid: 8 blocks (one per batch), 256 threads (thread = e)
// =====================================================================
__global__ void flow_q2c(const float* __restrict__ H,
                         const float* __restrict__ ws,
                         float* __restrict__ out) {
    __shared__ float sm[CTX];
    __shared__ float stmp;
    const int b   = blockIdx.x;
    const int tid = threadIdx.x;

    sm[tid]       = ws[WS_SMAX + (size_t)b * CTX + tid];
    sm[tid + 256] = ws[WS_SMAX + (size_t)b * CTX + tid + 256];
    __syncthreads();
    if (tid == 0) {
        float mx = -INFINITY;
        for (int c = 0; c < CTX; ++c) mx = fmaxf(mx, sm[c]);
        stmp = mx;
    }
    __syncthreads();
    const float mx = stmp;
    sm[tid]       = __expf(sm[tid] - mx);
    sm[tid + 256] = __expf(sm[tid + 256] - mx);
    __syncthreads();
    if (tid == 0) {
        float sum = 0.0f;
        for (int c = 0; c < CTX; ++c) sum += sm[c];
        stmp = 1.0f / sum;
    }
    __syncthreads();
    const float inv = stmp;

    float acc = 0.0f;
    for (int c = 0; c < CTX; ++c)
        acc = fmaf(sm[c], H[((size_t)b * CTX + c) * DIM + tid], acc);
    acc *= inv;

    for (int c = 0; c < CTX; ++c)
        out[((size_t)b * CTX + c) * DIM + tid] = acc;   // H_hat broadcast
}

// =====================================================================
extern "C" void kernel_launch(void* const* d_in, const int* in_sizes, int n_in,
                              void* d_out, int out_size, void* d_ws, size_t ws_size,
                              hipStream_t stream) {
    const float* H     = (const float*)d_in[0];
    const float* U     = (const float*)d_in[1];
    const int*   cmask = (const int*)d_in[2];
    const int*   qmask = (const int*)d_in[3];
    const float* W1    = (const float*)d_in[4];
    const float* b1    = (const float*)d_in[5];
    const float* W2    = (const float*)d_in[6];
    const float* b2    = (const float*)d_in[7];
    float* out = (float*)d_out;
    float* ws  = (float*)d_ws;

    // 1) hp / up precompute
    flow_precompute<<<BSZ * CTX + BSZ * QLEN, 256, 0, stream>>>(H, U, W1, b1, ws);

    // 2) fused WMMA attention core (228 KB dynamic LDS; WGP has 320 KB)
    dim3 grid(CTX / 16, BSZ);
    size_t smem = 131072 + 65536 + 16384 + 16384 + 4096;
    flow_main<<<grid, 256, smem, stream>>>(H, U, W1, W2, b2, cmask, qmask, ws, out);

    // 3) query-to-context attention + H_hat broadcast
    flow_q2c<<<BSZ, 256, 0, stream>>>(H, ws, out);
}
